// CeATTForTCPFormer_72550587564732
// MI455X (gfx1250) — matrix operand order repaired and checked by reading, not compiled, not run on hardware
//
#include <hip/hip_runtime.h>

// ---------------------------------------------------------------------------
// CDNA5 (gfx1250) wave32 WMMA implementation of CeATT-for-TCPFormer.
// All GEMMs use v_wmma_f32_16x16x32_bf16 (bf16 in, f32 accumulate).
// ---------------------------------------------------------------------------

typedef __attribute__((ext_vector_type(16))) __bf16 v16bf;
typedef __attribute__((ext_vector_type(8)))  float  v8f;

#define B_   32
#define T_   243
#define J_   17
#define C_   256
#define H_   8
#define DH_  32

#define NS_T 544                    // B*J temporal samples
#define LS_T 81                     // temporal seq len after stride-3 pool
#define TOK_T (NS_T * LS_T)         // 44064
#define NS_S 7776                   // B*T spatial samples
#define LS_S 8                      // spatial seq len after stride-2 pool
#define TOK_S (NS_S * LS_S)         // 62208
#define TOK_F (B_ * T_ * J_)        // 132192 fusion tokens

// ---------------------------------------------------------------------------
// helpers
// ---------------------------------------------------------------------------
__device__ __forceinline__ float gelu_exact(float x) {
    return 0.5f * x * (1.0f + erff(x * 0.70710678118654752f));
}

__device__ __forceinline__ v16bf zero_v16bf() {
    v16bf z;
#pragma unroll
    for (int i = 0; i < 16; ++i) z[i] = (__bf16)0.0f;
    return z;
}

__device__ __forceinline__ v8f zero_v8f() {
    v8f z;
#pragma unroll
    for (int i = 0; i < 8; ++i) z[i] = 0.0f;
    return z;
}

// A-matrix fragment (16x32 bf16 tile, row-major source, row stride = ldk).
// ISA layout: lane L -> row m = L&15, K chunks at kbase=(L>>4)*8 and kbase+16.
// `row` must be pre-clamped to a valid row (loads always legal).
__device__ __forceinline__ v16bf load_a_frag(const __bf16* __restrict__ A,
                                             long row, long ldk,
                                             int k0, int lane) {
    const __bf16* p = A + (size_t)row * ldk + k0 + ((lane >> 4) * 8);
    v16bf a;
#pragma unroll
    for (int i = 0; i < 8; ++i) { a[i] = p[i]; a[8 + i] = p[16 + i]; }
    return a;
}

// B-matrix fragment from row-major B^T (i.e. W stored N x K).
// ISA layout: lane L -> column n = L&15, K chunks at kbase=(L>>4)*8 and kbase+16.
__device__ __forceinline__ v16bf load_b_frag(const __bf16* __restrict__ W,
                                             long nrow, long ldk,
                                             int k0, int lane) {
    const __bf16* p = W + (size_t)nrow * ldk + k0 + ((lane >> 4) * 8);
    v16bf b;
#pragma unroll
    for (int i = 0; i < 8; ++i) { b[i] = p[i]; b[8 + i] = p[16 + i]; }
    return b;
}

// ---------------------------------------------------------------------------
// Generic GEMM:  out[M,N] = A[M,K](bf16) * W[N,K]^T(bf16) (+ bias)
// block: 256 thr = 8 waves (4 M-waves x 2 N-waves), wave tile 32x64,
// block tile 128x128. A rows clamped for loads; stores guarded on M.
// ---------------------------------------------------------------------------
__global__ __launch_bounds__(256) void gemm_bf16_wmma(
    const __bf16* __restrict__ A, const __bf16* __restrict__ W,
    const float* __restrict__ bias, void* __restrict__ out,
    long M, long N, long K, int out_is_bf16)
{
    const int lane = threadIdx.x & 31;
    const int wave = threadIdx.x >> 5;
    const int wm = wave >> 1;        // 0..3
    const int wn = wave & 1;         // 0..1
    const long m0 = (long)blockIdx.x * 128 + wm * 32;
    const long n0 = (long)blockIdx.y * 128 + wn * 64;

    v8f acc[2][4];
#pragma unroll
    for (int im = 0; im < 2; ++im)
#pragma unroll
        for (int in = 0; in < 4; ++in) acc[im][in] = zero_v8f();

    // clamped A rows (always-valid loads; garbage rows masked at store)
    long rowa[2];
#pragma unroll
    for (int im = 0; im < 2; ++im) {
        long r = m0 + im * 16 + (lane & 15);
        rowa[im] = (r < M) ? r : (M - 1);
    }

    for (int k0 = 0; k0 < K; k0 += 32) {
        v16bf af[2], bfr[4];
#pragma unroll
        for (int im = 0; im < 2; ++im)
            af[im] = load_a_frag(A, rowa[im], K, k0, lane);
#pragma unroll
        for (int in = 0; in < 4; ++in)
            bfr[in] = load_b_frag(W, n0 + in * 16 + (lane & 15), K, k0, lane);
#pragma unroll
        for (int im = 0; im < 2; ++im)
#pragma unroll
            for (int in = 0; in < 4; ++in)
                acc[im][in] = __builtin_amdgcn_wmma_f32_16x16x32_bf16(
                    false, af[im], false, bfr[in], (short)0, acc[im][in],
                    false, false);
    }

    // D layout: lane 0-15 col n=lane, rows r; lane 16-31 col n=lane-16, rows r+8
#pragma unroll
    for (int im = 0; im < 2; ++im) {
#pragma unroll
        for (int in = 0; in < 4; ++in) {
            const long col = n0 + in * 16 + (lane & 15);
            const float bv = bias ? bias[col] : 0.0f;
#pragma unroll
            for (int r = 0; r < 8; ++r) {
                const long row = m0 + im * 16 + ((lane >> 4) * 8) + r;
                if (row < M) {
                    const float v = acc[im][in][r] + bv;
                    if (out_is_bf16)
                        ((__bf16*)out)[(size_t)row * N + col] = (__bf16)v;
                    else
                        ((float*)out)[(size_t)row * N + col] = v;
                }
            }
        }
    }
}

// ---------------------------------------------------------------------------
// WMMA multi-head attention.
// One block per (sample n, head h); one wave per 16-query M-tile.
// qkv: token-major (n*Ls + l) x 768 bf16, q at +0, k at +256, v at +512
// out: token-major (n*Ls + l) x 256 bf16
// ---------------------------------------------------------------------------
#define MAX_LT 96
#define MAX_K2 96

__global__ void attn_wmma(const __bf16* __restrict__ qkv,
                          __bf16* __restrict__ outp,
                          int Ls, int numTiles, int K2pad, int k2chunks)
{
    __shared__ __bf16 Pld[MAX_LT * MAX_K2];   // softmax probs, row-major
    __shared__ __bf16 Vt[32 * MAX_K2];        // v transposed: [d][l]

    const int lane = threadIdx.x & 31;
    const int mt   = threadIdx.x >> 5;        // wave id == query tile
    const int n    = blockIdx.x >> 3;
    const int h    = blockIdx.x & 7;
    const int LT   = numTiles * 16;

    const __bf16* base = qkv + (size_t)n * Ls * 768 + h * 32;

    // zero LDS (covers softmax/K padding)
    const int tot = LT * K2pad + 32 * K2pad;
    for (int i = threadIdx.x; i < tot; i += blockDim.x) {
        if (i < LT * K2pad) Pld[i] = (__bf16)0.0f;
        else                Vt[i - LT * K2pad] = (__bf16)0.0f;
    }
    // stage v^T into LDS
    for (int i = threadIdx.x; i < Ls * 32; i += blockDim.x) {
        const int l = i >> 5, d = i & 31;
        Vt[d * K2pad + l] = base[(size_t)l * 768 + 512 + d];
    }

    // ---- phase 1: S = q k^T  (A = q 16x32, B = k^T via row-major k) ----
    v16bf aq;
    {
        const int l = mt * 16 + (lane & 15);
        if (l < Ls) {
            const __bf16* p = base + (size_t)l * 768 + ((lane >> 4) * 8);
#pragma unroll
            for (int i = 0; i < 8; ++i) { aq[i] = p[i]; aq[8 + i] = p[16 + i]; }
        } else {
            aq = zero_v16bf();
        }
    }

    v8f S[6];
    for (int nt = 0; nt < numTiles; ++nt) {
        v16bf bk;
        const int kk = nt * 16 + (lane & 15);
        if (kk < Ls) {
            const __bf16* p = base + (size_t)kk * 768 + 256 + ((lane >> 4) * 8);
#pragma unroll
            for (int i = 0; i < 8; ++i) { bk[i] = p[i]; bk[8 + i] = p[16 + i]; }
        } else {
            bk = zero_v16bf();
        }
        v8f z = zero_v8f();
        S[nt] = __builtin_amdgcn_wmma_f32_16x16x32_bf16(
            false, aq, false, bk, (short)0, z, false, false);
    }

    // ---- softmax over keys (row = 16 lanes of one half-wave) ----
    const float scale = 0.17677669529663687f;   // 1/sqrt(32)
#pragma unroll
    for (int r = 0; r < 8; ++r) {
        float m = -3.0e38f;
        for (int nt = 0; nt < numTiles; ++nt) {
            const int col = nt * 16 + (lane & 15);
            const float v = (col < Ls) ? S[nt][r] * scale : -3.0e38f;
            S[nt][r] = v;
            m = fmaxf(m, v);
        }
#pragma unroll
        for (int msk = 1; msk <= 8; msk <<= 1)
            m = fmaxf(m, __shfl_xor(m, msk, 32));
        float s = 0.0f;
        for (int nt = 0; nt < numTiles; ++nt) {
            const float e = __expf(S[nt][r] - m);
            S[nt][r] = e;
            s += e;
        }
#pragma unroll
        for (int msk = 1; msk <= 8; msk <<= 1)
            s += __shfl_xor(s, msk, 32);
        const float inv = 1.0f / s;
        for (int nt = 0; nt < numTiles; ++nt) S[nt][r] *= inv;
    }

    // write P tiles to LDS (bf16)
    for (int nt = 0; nt < numTiles; ++nt) {
#pragma unroll
        for (int r = 0; r < 8; ++r) {
            const int row = mt * 16 + ((lane >> 4) * 8) + r;
            const int col = nt * 16 + (lane & 15);
            Pld[row * K2pad + col] = (__bf16)S[nt][r];
        }
    }
    __syncthreads();

    // ---- phase 2: O = P v  (A = P from LDS, B = v^T from LDS) ----
    for (int n2 = 0; n2 < 2; ++n2) {
        v8f o = zero_v8f();
        for (int kc = 0; kc < k2chunks; ++kc) {
            v16bf pa;
            {
                const int row = mt * 16 + (lane & 15);
                const __bf16* p = &Pld[row * K2pad + kc * 32 + ((lane >> 4) * 8)];
#pragma unroll
                for (int i = 0; i < 8; ++i) { pa[i] = p[i]; pa[8 + i] = p[16 + i]; }
            }
            v16bf vb;
            {
                const int d = n2 * 16 + (lane & 15);
                const __bf16* p = &Vt[d * K2pad + kc * 32 + ((lane >> 4) * 8)];
#pragma unroll
                for (int i = 0; i < 8; ++i) { vb[i] = p[i]; vb[8 + i] = p[16 + i]; }
            }
            o = __builtin_amdgcn_wmma_f32_16x16x32_bf16(
                false, pa, false, vb, (short)0, o, false, false);
        }
#pragma unroll
        for (int r = 0; r < 8; ++r) {
            const int l = mt * 16 + ((lane >> 4) * 8) + r;
            if (l < Ls) {
                const int d = n2 * 16 + (lane & 15);
                outp[((size_t)(n * Ls + l)) * 256 + h * 32 + d] = (__bf16)o[r];
            }
        }
    }
}

// ---------------------------------------------------------------------------
// elementwise / reduction kernels
// ---------------------------------------------------------------------------
__global__ void w2bf_kernel(const float* __restrict__ s, __bf16* __restrict__ d,
                            long n) {
    long i = (long)blockIdx.x * blockDim.x + threadIdx.x;
    if (i < n) d[i] = (__bf16)s[i];
}

__global__ void zero_f32(float* __restrict__ p, int n) {
    int i = blockIdx.x * blockDim.x + threadIdx.x;
    if (i < n) p[i] = 0.0f;
}

// temporal prep: stride-3 mean pool of x (B,T,J,C) -> (B*J, 81, C) bf16
__global__ void prep_temporal(const float* __restrict__ x,
                              __bf16* __restrict__ xt) {
    const long total = (long)TOK_T * 256;
    long i = (long)blockIdx.x * blockDim.x + threadIdx.x;
    if (i >= total) return;
    const int c = (int)(i & 255);
    const long tok = i >> 8;
    const int l = (int)(tok % LS_T);
    const int nn = (int)(tok / LS_T);
    const int b = nn / J_, j = nn % J_;
    const size_t xb = (((size_t)b * T_ + 3 * l) * J_ + j) * 256 + c;
    const float v = (x[xb] + x[xb + (size_t)J_ * 256] + x[xb + 2 * (size_t)J_ * 256])
                    * (1.0f / 3.0f);
    xt[i] = (__bf16)v;
}

// spatial prep: stride-2 mean pool of x (B,T,J,C) over J -> (B*T, 8, C) bf16
__global__ void prep_spatial(const float* __restrict__ x,
                             __bf16* __restrict__ xs) {
    const long total = (long)TOK_S * 256;
    long i = (long)blockIdx.x * blockDim.x + threadIdx.x;
    if (i >= total) return;
    const int c = (int)(i & 255);
    const long tok = i >> 8;
    const int l = (int)(tok & 7);
    const long nn = tok >> 3;                    // b*T + t
    const size_t xb = ((size_t)nn * J_ + 2 * l) * 256 + c;
    xs[i] = (__bf16)(0.5f * (x[xb] + x[xb + 256]));
}

// depthwise conv (k=3, pad=1) along seq + BN partial sums. block = one sample.
__global__ void dwconv_bnstat(const float* __restrict__ y,
                              const float* __restrict__ dw,
                              float* __restrict__ co,
                              float* __restrict__ bnsum,
                              float* __restrict__ bnsumsq, int Ls) {
    const int n = blockIdx.x;
    const int c = threadIdx.x;                   // 256 threads
    const float w0 = dw[c * 3 + 0], w1 = dw[c * 3 + 1], w2 = dw[c * 3 + 2];
    const float* yb = y + (size_t)n * Ls * 256 + c;
    float* cb = co + (size_t)n * Ls * 256 + c;
    float s = 0.0f, ss = 0.0f;
    for (int l = 0; l < Ls; ++l) {
        const float xm = (l > 0)      ? yb[(size_t)(l - 1) * 256] : 0.0f;
        const float x0 = yb[(size_t)l * 256];
        const float xp = (l < Ls - 1) ? yb[(size_t)(l + 1) * 256] : 0.0f;
        const float v = w0 * xm + w1 * x0 + w2 * xp;
        cb[(size_t)l * 256] = v;
        s += v; ss += v * v;
    }
    atomicAdd(&bnsum[c], s);
    atomicAdd(&bnsumsq[c], ss);
}

__global__ void bn_gelu(const float* __restrict__ co,
                        const float* __restrict__ bnsum,
                        const float* __restrict__ bnsumsq,
                        const float* __restrict__ g, const float* __restrict__ b,
                        __bf16* __restrict__ act, long total, float inv_count) {
    long i = (long)blockIdx.x * blockDim.x + threadIdx.x;
    if (i >= total) return;
    const int c = (int)(i & 255);
    const float mean = bnsum[c] * inv_count;
    const float var = bnsumsq[c] * inv_count - mean * mean;
    float v = (co[i] - mean) * rsqrtf(var + 1e-5f) * g[c] + b[c];
    act[i] = (__bf16)gelu_exact(v);
}

// temporal interp 81 -> 243, write bf16 into comb cols [0,256)
__global__ void interp_temporal(const float* __restrict__ ypw,
                                __bf16* __restrict__ comb) {
    const long total = (long)NS_T * T_ * 256;
    long i = (long)blockIdx.x * blockDim.x + threadIdx.x;
    if (i >= total) return;
    const int c = (int)(i & 255);
    const long r = i >> 8;
    const int t = (int)(r % T_);
    const int nn = (int)(r / T_);
    const int b = nn / J_, j = nn % J_;
    float coord = (t + 0.5f) * (1.0f / 3.0f) - 0.5f;
    coord = fminf(fmaxf(coord, 0.0f), (float)(LS_T - 1));
    const int lo = (int)floorf(coord);
    const int hi = min(lo + 1, LS_T - 1);
    const float w = coord - (float)lo;
    const float* yb = ypw + ((size_t)nn * LS_T) * 256 + c;
    const float v = yb[(size_t)lo * 256] * (1.0f - w) + yb[(size_t)hi * 256] * w;
    comb[(((size_t)b * T_ + t) * J_ + j) * 512 + c] = (__bf16)v;
}

// spatial interp 8 -> 17, write bf16 into comb cols [256,512)
__global__ void interp_spatial(const float* __restrict__ ypw,
                               __bf16* __restrict__ comb) {
    const long total = (long)NS_S * J_ * 256;
    long i = (long)blockIdx.x * blockDim.x + threadIdx.x;
    if (i >= total) return;
    const int c = (int)(i & 255);
    const long r = i >> 8;
    const int j = (int)(r % J_);
    const long nn = r / J_;                      // b*T + t
    float coord = (j + 0.5f) * (8.0f / 17.0f) - 0.5f;
    coord = fminf(fmaxf(coord, 0.0f), (float)(LS_S - 1));
    const int lo = (int)floorf(coord);
    const int hi = min(lo + 1, LS_S - 1);
    const float w = coord - (float)lo;
    const float* yb = ypw + ((size_t)nn * LS_S) * 256 + c;
    const float v = yb[(size_t)lo * 256] * (1.0f - w) + yb[(size_t)hi * 256] * w;
    comb[((size_t)nn * J_ + j) * 512 + 256 + c] = (__bf16)v;
}

// LayerNorm(C=256) + GELU, one wave per token (8 ch/lane), -> bf16
__global__ void ln_gelu(const float* __restrict__ h,
                        const float* __restrict__ g, const float* __restrict__ b,
                        __bf16* __restrict__ act, int tokens) {
    const int warp = threadIdx.x >> 5;
    const int lane = threadIdx.x & 31;
    const int tok = blockIdx.x * 8 + warp;
    if (tok >= tokens) return;
    const float* p = h + (size_t)tok * 256;
    float vals[8];
    float s = 0.0f;
#pragma unroll
    for (int q = 0; q < 8; ++q) { vals[q] = p[lane + q * 32]; s += vals[q]; }
#pragma unroll
    for (int msk = 1; msk <= 16; msk <<= 1) s += __shfl_xor(s, msk, 32);
    const float mean = s * (1.0f / 256.0f);
    float ss = 0.0f;
#pragma unroll
    for (int q = 0; q < 8; ++q) { const float d = vals[q] - mean; ss += d * d; }
#pragma unroll
    for (int msk = 1; msk <= 16; msk <<= 1) ss += __shfl_xor(ss, msk, 32);
    const float inv = rsqrtf(ss * (1.0f / 256.0f) + 1e-5f);
    __bf16* o = act + (size_t)tok * 256;
#pragma unroll
    for (int q = 0; q < 8; ++q) {
        const int c = lane + q * 32;
        const float v = (vals[q] - mean) * inv * g[c] + b[c];
        o[c] = (__bf16)gelu_exact(v);
    }
}

// ---------------------------------------------------------------------------
// host-side launch
// ---------------------------------------------------------------------------
static inline size_t alignup(size_t x) { return (x + 255) & ~(size_t)255; }
static inline int cdivi(long a, long b) { return (int)((a + b - 1) / b); }

extern "C" void kernel_launch(void* const* d_in, const int* in_sizes, int n_in,
                              void* d_out, int out_size, void* d_ws,
                              size_t ws_size, hipStream_t stream) {
    (void)in_sizes; (void)n_in; (void)out_size; (void)ws_size;

    const float* x      = (const float*)d_in[0];
    const float* t_qkv  = (const float*)d_in[1];
    const float* t_proj = (const float*)d_in[2];
    const float* t_dw   = (const float*)d_in[3];
    const float* t_bn_g = (const float*)d_in[4];
    const float* t_bn_b = (const float*)d_in[5];
    const float* t_pw   = (const float*)d_in[6];
    const float* s_qkv  = (const float*)d_in[7];
    const float* s_proj = (const float*)d_in[8];
    const float* s_dw   = (const float*)d_in[9];
    const float* s_bn_g = (const float*)d_in[10];
    const float* s_bn_b = (const float*)d_in[11];
    const float* s_pw   = (const float*)d_in[12];
    const float* f_w1   = (const float*)d_in[13];
    const float* f_b1   = (const float*)d_in[14];
    const float* f_ln_g = (const float*)d_in[15];
    const float* f_ln_b = (const float*)d_in[16];
    const float* f_w2   = (const float*)d_in[17];
    const float* f_b2   = (const float*)d_in[18];
    const float* p_w    = (const float*)d_in[19];
    const float* p_b    = (const float*)d_in[20];

    char* ws = (char*)d_ws;
    size_t off = 0;
    auto carve = [&](size_t bytes) -> void* {
        void* p = ws + off; off = alignup(off + bytes); return p;
    };

    // bf16 weights (persistent)
    __bf16* w_tqkv  = (__bf16*)carve((size_t)768 * 256 * 2);
    __bf16* w_tproj = (__bf16*)carve((size_t)256 * 256 * 2);
    __bf16* w_tpw   = (__bf16*)carve((size_t)256 * 256 * 2);
    __bf16* w_sqkv  = (__bf16*)carve((size_t)768 * 256 * 2);
    __bf16* w_sproj = (__bf16*)carve((size_t)256 * 256 * 2);
    __bf16* w_spw   = (__bf16*)carve((size_t)256 * 256 * 2);
    __bf16* w_fw1   = (__bf16*)carve((size_t)256 * 512 * 2);
    __bf16* w_fw2   = (__bf16*)carve((size_t)256 * 256 * 2);
    __bf16* w_pw    = (__bf16*)carve((size_t)256 * 256 * 2);
    float*  bnsum   = (float*)carve(256 * 4);
    float*  bnsumsq = (float*)carve(256 * 4);
    __bf16* comb    = (__bf16*)carve((size_t)TOK_F * 512 * 2);

    // branch arena (sized for spatial = larger branch), reused by fusion head
    const size_t arena = off;
    size_t ao = arena;
    __bf16* bufA   = (__bf16*)(ws + ao); ao = alignup(ao + (size_t)TOK_S * 256 * 2);
    __bf16* bufQKV = (__bf16*)(ws + ao); ao = alignup(ao + (size_t)TOK_S * 768 * 2);
    __bf16* bufATT = (__bf16*)(ws + ao); ao = alignup(ao + (size_t)TOK_S * 256 * 2);
    float*  bufPROJ= (float*) (ws + ao); ao = alignup(ao + (size_t)TOK_S * 256 * 4);
    __bf16* bufACT = (__bf16*)(ws + ao); ao = alignup(ao + (size_t)TOK_S * 256 * 2);
    float*  bufPW  = (float*) (ws + ao); ao = alignup(ao + (size_t)TOK_S * 256 * 4);
    // fusion-head buffers overlay the (then dead) branch buffers
    size_t fo = arena;
    float*  hbuf = (float*)(ws + fo);  fo = alignup(fo + (size_t)TOK_F * 256 * 4);
    __bf16* actf = (__bf16*)(ws + fo); fo = alignup(fo + (size_t)TOK_F * 256 * 2);
    __bf16* y3   = (__bf16*)(ws + fo); fo = alignup(fo + (size_t)TOK_F * 256 * 2);

    auto conv_w = [&](const float* s, __bf16* d, long n) {
        w2bf_kernel<<<cdivi(n, 256), 256, 0, stream>>>(s, d, n);
    };
    auto gemm = [&](const __bf16* A, const __bf16* W, const float* bias,
                    void* out, long M, long N, long K, int out_bf16) {
        dim3 grid(cdivi(M, 128), (unsigned)(N / 128));
        gemm_bf16_wmma<<<grid, 256, 0, stream>>>(A, W, bias, out, M, N, K,
                                                 out_bf16);
    };

    // ---- weights -> bf16 ----
    conv_w(t_qkv,  w_tqkv,  768 * 256);
    conv_w(t_proj, w_tproj, 256 * 256);
    conv_w(t_pw,   w_tpw,   256 * 256);
    conv_w(s_qkv,  w_sqkv,  768 * 256);
    conv_w(s_proj, w_sproj, 256 * 256);
    conv_w(s_pw,   w_spw,   256 * 256);
    conv_w(f_w1,   w_fw1,   256 * 512);
    conv_w(f_w2,   w_fw2,   256 * 256);
    conv_w(p_w,    w_pw,    256 * 256);

    // ================= temporal branch =================
    prep_temporal<<<cdivi((long)TOK_T * 256, 256), 256, 0, stream>>>(x, bufA);
    gemm(bufA, w_tqkv, nullptr, bufQKV, TOK_T, 768, 256, 1);
    attn_wmma<<<NS_T * H_, 6 * 32, 0, stream>>>(bufQKV, bufATT, LS_T, 6, 96, 3);
    gemm(bufATT, w_tproj, nullptr, bufPROJ, TOK_T, 256, 256, 0);
    zero_f32<<<2, 256, 0, stream>>>(bnsum, 256);
    zero_f32<<<2, 256, 0, stream>>>(bnsumsq, 256);
    dwconv_bnstat<<<NS_T, 256, 0, stream>>>(bufPROJ, t_dw, bufPW, bnsum,
                                            bnsumsq, LS_T);
    bn_gelu<<<cdivi((long)TOK_T * 256, 256), 256, 0, stream>>>(
        bufPW, bnsum, bnsumsq, t_bn_g, t_bn_b, bufACT, (long)TOK_T * 256,
        1.0f / (float)TOK_T);
    gemm(bufACT, w_tpw, nullptr, bufPW, TOK_T, 256, 256, 0);
    interp_temporal<<<cdivi((long)NS_T * T_ * 256, 256), 256, 0, stream>>>(
        bufPW, comb);

    // ================= spatial branch =================
    prep_spatial<<<cdivi((long)TOK_S * 256, 256), 256, 0, stream>>>(x, bufA);
    gemm(bufA, w_sqkv, nullptr, bufQKV, TOK_S, 768, 256, 1);
    attn_wmma<<<NS_S * H_, 32, 0, stream>>>(bufQKV, bufATT, LS_S, 1, 32, 1);
    gemm(bufATT, w_sproj, nullptr, bufPROJ, TOK_S, 256, 256, 0);
    zero_f32<<<2, 256, 0, stream>>>(bnsum, 256);
    zero_f32<<<2, 256, 0, stream>>>(bnsumsq, 256);
    dwconv_bnstat<<<NS_S, 256, 0, stream>>>(bufPROJ, s_dw, bufPW, bnsum,
                                            bnsumsq, LS_S);
    bn_gelu<<<cdivi((long)TOK_S * 256, 256), 256, 0, stream>>>(
        bufPW, bnsum, bnsumsq, s_bn_g, s_bn_b, bufACT, (long)TOK_S * 256,
        1.0f / (float)TOK_S);
    gemm(bufACT, w_spw, nullptr, bufPW, TOK_S, 256, 256, 0);
    interp_spatial<<<cdivi((long)NS_S * J_ * 256, 256), 256, 0, stream>>>(
        bufPW, comb);

    // ================= fusion head =================
    gemm(comb, w_fw1, f_b1, hbuf, TOK_F, 256, 512, 0);
    ln_gelu<<<cdivi(TOK_F, 8), 256, 0, stream>>>(hbuf, f_ln_g, f_ln_b, actf,
                                                 TOK_F);
    gemm(actf, w_fw2, f_b2, y3, TOK_F, 256, 256, 1);
    gemm(y3, w_pw, p_b, (float*)d_out, TOK_F, 256, 256, 0);
}
